// MixtralExpertCapacityTop2MLP_28286654611540
// MI455X (gfx1250) — matrix operand order repaired and checked by reading, not compiled, not run on hardware
//
#include <hip/hip_runtime.h>
#include <stdint.h>

// ---- problem sizes (E, B, C, H, F) = (8, 4, 512, 2048, 7168) ----
constexpr int NE  = 8;
constexpr int TOK = 4 * 512;   // B*C tokens per expert
constexpr int HID = 2048;      // H
constexpr int FFN = 7168;      // F

// ---- tiling ----
constexpr int BM = 128;
constexpr int BN = 128;
constexpr int BK = 32;                  // one WMMA K-step per stage
constexpr int LDSTR = BK + 8;           // 40 bf16 -> 80B row stride (16B aligned)
constexpr int TILE_U16 = BM * LDSTR;    // 5120 elems = 10240 B per buffer

typedef __attribute__((ext_vector_type(16))) __bf16 v16bf;
typedef __attribute__((ext_vector_type(8)))  float  v8f;

union FragU { v16bf v; uint4 q[2]; };

__device__ __forceinline__ uint16_t f2bf_rne(float x) {
    uint32_t u = __float_as_uint(x);
    uint32_t r = u + 0x7FFFu + ((u >> 16) & 1u);
    return (uint16_t)(r >> 16);
}

// low 32 bits of a flat shared-aperture address = LDS byte address (ISA 10.2)
__device__ __forceinline__ uint32_t lds_off32(const void* p) {
    return (uint32_t)(uintptr_t)p;
}

// async global->LDS 16B copy, per-lane addresses, tracked with ASYNCcnt
__device__ __forceinline__ void async_b128(uint32_t lds_byte, const uint16_t* g) {
    asm volatile("global_load_async_to_lds_b128 %0, %1, off"
                 :: "v"(lds_byte), "v"(g) : "memory");
}

__device__ __forceinline__ void wait_async0() {
#if __has_builtin(__builtin_amdgcn_s_wait_asynccnt)
    __builtin_amdgcn_s_wait_asynccnt(0);
#else
    asm volatile("s_wait_asynccnt 0x0" ::: "memory");
#endif
}

// A fragment (16x32 bf16, LDS tile [m][k], stride LDSTR):
// lanes 0-15: row m+lane,    K {0..7, 16..23}
// lanes16-31: row m+lane-16, K {8..15, 24..31}
__device__ __forceinline__ v16bf load_a_frag(const uint16_t* a, int m, int lane) {
    int row = m + (lane & 15);
    int klo = (lane >> 4) << 3;
    const uint4* p = reinterpret_cast<const uint4*>(a + row * LDSTR + klo);
    FragU f;
    f.q[0] = p[0];   // K +0..7
    f.q[1] = p[2];   // K +16..23 (+32 bytes)
    return f.v;
}

// B fragment (32x16 bf16), LDS tile [n][k], stride LDSTR:
// lanes 0-15: col n+lane, K {0..7, 16..23} ; lanes 16-31: col n+lane-16, K {8..15, 24..31}
__device__ __forceinline__ v16bf load_b_frag(const uint16_t* b, int n, int lane) {
    int col = n + (lane & 15);
    int ks  = (lane >> 4) << 3;
    const uint4* p = reinterpret_cast<const uint4*>(b + col * LDSTR + ks);
    FragU f;
    f.q[0] = p[0];
    f.q[1] = p[2];
    return f.v;
}

__device__ __forceinline__ v8f wmma_bf16(v16bf a, v16bf b, v8f c) {
    return __builtin_amdgcn_wmma_f32_16x16x32_bf16(false, a, false, b, (short)0, c, false, false);
}

// ============================================================================
// One-time fp32 -> bf16 convert (straight copy layout)
// ============================================================================
__global__ __launch_bounds__(256) void convert_bf16(
    const float* __restrict__ in, uint16_t* __restrict__ out, int n4)
{
    int i = blockIdx.x * 256 + threadIdx.x;
    if (i >= n4) return;
    float4 v = reinterpret_cast<const float4*>(in)[i];
    uint2 o;
    o.x = ((uint32_t)f2bf_rne(v.y) << 16) | f2bf_rne(v.x);
    o.y = ((uint32_t)f2bf_rne(v.w) << 16) | f2bf_rne(v.z);
    reinterpret_cast<uint2*>(out)[i] = o;
}

// ============================================================================
// One-time fp32 -> bf16 convert + transpose:  in (E,R,C) -> out (E,C,R)
// 32x32 tiles through LDS, coalesced both sides
// ============================================================================
__global__ __launch_bounds__(256) void convert_transpose_bf16(
    const float* __restrict__ in, uint16_t* __restrict__ out, int R, int C)
{
    __shared__ uint16_t t[32][33];
    const int e  = blockIdx.z;
    const int r0 = blockIdx.y * 32;
    const int c0 = blockIdx.x * 32;
    const float* ie = in  + (size_t)e * R * C;
    uint16_t*    oe = out + (size_t)e * R * C;
    const int tid = threadIdx.x;
    {
        int r = tid >> 3, c4 = (tid & 7) * 4;
        float4 v = *reinterpret_cast<const float4*>(ie + (size_t)(r0 + r) * C + c0 + c4);
        t[r][c4 + 0] = f2bf_rne(v.x);
        t[r][c4 + 1] = f2bf_rne(v.y);
        t[r][c4 + 2] = f2bf_rne(v.z);
        t[r][c4 + 3] = f2bf_rne(v.w);
    }
    __syncthreads();
    {
        int c = tid >> 3, r4 = (tid & 7) * 4;
        uint2 o;
        o.x = ((uint32_t)t[r4 + 1][c] << 16) | t[r4 + 0][c];
        o.y = ((uint32_t)t[r4 + 3][c] << 16) | t[r4 + 2][c];
        *reinterpret_cast<uint2*>(oe + (size_t)(c0 + c) * R + r0 + r4) = o;
    }
}

// ============================================================================
// Kernel 1: hidden = silu(X @ W1) * (X @ W3)   (all-bf16 operands, f32 accum)
//   Xb  : (E, TOK, HID)   W1t/W3t : (E, FFN, HID)  [pre-transposed]
//   Hws : (E, TOK, FFN) bf16
// Double-buffered LDS, async global->LDS staging.
// ============================================================================
__global__ __launch_bounds__(256) void moe_gemm1_swiglu(
    const uint16_t* __restrict__ Xb, const uint16_t* __restrict__ W1t,
    const uint16_t* __restrict__ W3t, uint16_t* __restrict__ Hws)
{
    __shared__ __align__(16) uint16_t As [2][TILE_U16];
    __shared__ __align__(16) uint16_t B1s[2][TILE_U16];
    __shared__ __align__(16) uint16_t B3s[2][TILE_U16];

    const int e  = blockIdx.z;
    const int m0 = blockIdx.y * BM;
    const int n0 = blockIdx.x * BN;
    const int tid = threadIdx.x, lane = tid & 31, wave = tid >> 5;
    const int wm = (wave & 3) * 32, wn = (wave >> 2) * 64;

    const uint16_t* Xe  = Xb  + (size_t)e * TOK * HID + (size_t)m0 * HID;
    const uint16_t* W1e = W1t + (size_t)e * FFN * HID + (size_t)n0 * HID;
    const uint16_t* W3e = W3t + (size_t)e * FFN * HID + (size_t)n0 * HID;

    const uint32_t aL[2]  = { lds_off32(&As [0][0]), lds_off32(&As [1][0]) };
    const uint32_t b1L[2] = { lds_off32(&B1s[0][0]), lds_off32(&B1s[1][0]) };
    const uint32_t b3L[2] = { lds_off32(&B3s[0][0]), lds_off32(&B3s[1][0]) };

    // staging: 128 rows x 32 bf16 per tile = 512 uint4 -> 2 per thread per matrix
    const int srow0 = tid >> 2;             // q = tid    : rows 0..63
    const int srow1 = (tid + 256) >> 2;     // q = tid+256: rows 64..127
    const int sc8   = (tid & 3) * 8;
    const uint32_t sLoff0 = (uint32_t)((srow0 * LDSTR + sc8) * 2);
    const uint32_t sLoff1 = (uint32_t)((srow1 * LDSTR + sc8) * 2);

    v8f accG[2][4], accU[2][4];
    #pragma unroll
    for (int i = 0; i < 2; ++i)
        #pragma unroll
        for (int j = 0; j < 4; ++j) { accG[i][j] = {}; accU[i][j] = {}; }

    // prologue: stage tile 0 into buffer 0
    {
        async_b128(aL[0]  + sLoff0, Xe  + (size_t)srow0 * HID + sc8);
        async_b128(aL[0]  + sLoff1, Xe  + (size_t)srow1 * HID + sc8);
        async_b128(b1L[0] + sLoff0, W1e + (size_t)srow0 * HID + sc8);
        async_b128(b1L[0] + sLoff1, W1e + (size_t)srow1 * HID + sc8);
        async_b128(b3L[0] + sLoff0, W3e + (size_t)srow0 * HID + sc8);
        async_b128(b3L[0] + sLoff1, W3e + (size_t)srow1 * HID + sc8);
    }

    constexpr int NIT = HID / BK;   // 64
    for (int it = 0; it < NIT; ++it) {
        const int cur = it & 1;
        wait_async0();        // this wave's stage of tile `it` has landed
        __syncthreads();      // all waves' stages landed; prior reads of nxt done
        if (it + 1 < NIT) {
            const int nxt = cur ^ 1;
            const int k0  = (it + 1) * BK;
            async_b128(aL[nxt]  + sLoff0, Xe  + (size_t)srow0 * HID + k0 + sc8);
            async_b128(aL[nxt]  + sLoff1, Xe  + (size_t)srow1 * HID + k0 + sc8);
            async_b128(b1L[nxt] + sLoff0, W1e + (size_t)srow0 * HID + k0 + sc8);
            async_b128(b1L[nxt] + sLoff1, W1e + (size_t)srow1 * HID + k0 + sc8);
            async_b128(b3L[nxt] + sLoff0, W3e + (size_t)srow0 * HID + k0 + sc8);
            async_b128(b3L[nxt] + sLoff1, W3e + (size_t)srow1 * HID + k0 + sc8);
        }
        const uint16_t* A  = As [cur];
        const uint16_t* B1 = B1s[cur];
        const uint16_t* B3 = B3s[cur];

        v16bf a0 = load_a_frag(A, wm, lane);
        v16bf a1 = load_a_frag(A, wm + 16, lane);
        v16bf b1f[4], b3f[4];
        #pragma unroll
        for (int nj = 0; nj < 4; ++nj) {
            b1f[nj] = load_b_frag(B1, wn + nj * 16, lane);
            b3f[nj] = load_b_frag(B3, wn + nj * 16, lane);
        }
        #pragma unroll
        for (int nj = 0; nj < 4; ++nj) {
            accG[0][nj] = wmma_bf16(a0, b1f[nj], accG[0][nj]);
            accG[1][nj] = wmma_bf16(a1, b1f[nj], accG[1][nj]);
            accU[0][nj] = wmma_bf16(a0, b3f[nj], accU[0][nj]);
            accU[1][nj] = wmma_bf16(a1, b3f[nj], accU[1][nj]);
        }
    }

    // ---- epilogue: silu(gate) * up -> bf16 ----
    uint16_t* Ho = Hws + (size_t)e * TOK * FFN;
    const int mlane = (lane >> 4) << 3;   // 0 or 8
    const int nl    = lane & 15;
    #pragma unroll
    for (int mi = 0; mi < 2; ++mi)
        #pragma unroll
        for (int nj = 0; nj < 4; ++nj)
            #pragma unroll
            for (int r = 0; r < 8; ++r) {
                float g = accG[mi][nj][r];
                float u = accU[mi][nj][r];
                float h = (g / (1.0f + __expf(-g))) * u;
                int m = m0 + wm + mi * 16 + mlane + r;
                int n = n0 + wn + nj * 16 + nl;
                Ho[(size_t)m * FFN + n] = f2bf_rne(h);
            }
}

// ============================================================================
// Kernel 2: out = hidden @ W2    (bf16 x bf16, fp32 out)
//   Hws : (E, TOK, FFN) bf16 ;  W2t : (E, HID, FFN) bf16 [pre-transposed]
// ============================================================================
__global__ __launch_bounds__(256) void moe_gemm2(
    const uint16_t* __restrict__ Hws, const uint16_t* __restrict__ W2t,
    float* __restrict__ Out)
{
    __shared__ __align__(16) uint16_t As[2][TILE_U16];
    __shared__ __align__(16) uint16_t Bs[2][TILE_U16];

    const int e  = blockIdx.z;
    const int m0 = blockIdx.y * BM;
    const int n0 = blockIdx.x * BN;
    const int tid = threadIdx.x, lane = tid & 31, wave = tid >> 5;
    const int wm = (wave & 3) * 32, wn = (wave >> 2) * 64;

    const uint16_t* He  = Hws + (size_t)e * TOK * FFN + (size_t)m0 * FFN;
    const uint16_t* W2e = W2t + (size_t)e * HID * FFN + (size_t)n0 * FFN;

    const uint32_t aL[2] = { lds_off32(&As[0][0]), lds_off32(&As[1][0]) };
    const uint32_t bL[2] = { lds_off32(&Bs[0][0]), lds_off32(&Bs[1][0]) };

    const int srow0 = tid >> 2;
    const int srow1 = (tid + 256) >> 2;
    const int sc8   = (tid & 3) * 8;
    const uint32_t sLoff0 = (uint32_t)((srow0 * LDSTR + sc8) * 2);
    const uint32_t sLoff1 = (uint32_t)((srow1 * LDSTR + sc8) * 2);

    v8f acc[2][4];
    #pragma unroll
    for (int i = 0; i < 2; ++i)
        #pragma unroll
        for (int j = 0; j < 4; ++j) acc[i][j] = {};

    {
        async_b128(aL[0] + sLoff0, He  + (size_t)srow0 * FFN + sc8);
        async_b128(aL[0] + sLoff1, He  + (size_t)srow1 * FFN + sc8);
        async_b128(bL[0] + sLoff0, W2e + (size_t)srow0 * FFN + sc8);
        async_b128(bL[0] + sLoff1, W2e + (size_t)srow1 * FFN + sc8);
    }

    constexpr int NIT = FFN / BK;   // 224
    for (int it = 0; it < NIT; ++it) {
        const int cur = it & 1;
        wait_async0();
        __syncthreads();
        if (it + 1 < NIT) {
            const int nxt = cur ^ 1;
            const int k0  = (it + 1) * BK;
            async_b128(aL[nxt] + sLoff0, He  + (size_t)srow0 * FFN + k0 + sc8);
            async_b128(aL[nxt] + sLoff1, He  + (size_t)srow1 * FFN + k0 + sc8);
            async_b128(bL[nxt] + sLoff0, W2e + (size_t)srow0 * FFN + k0 + sc8);
            async_b128(bL[nxt] + sLoff1, W2e + (size_t)srow1 * FFN + k0 + sc8);
        }
        const uint16_t* A = As[cur];
        const uint16_t* B = Bs[cur];

        v16bf a0 = load_a_frag(A, wm, lane);
        v16bf a1 = load_a_frag(A, wm + 16, lane);
        v16bf bf[4];
        #pragma unroll
        for (int nj = 0; nj < 4; ++nj) bf[nj] = load_b_frag(B, wn + nj * 16, lane);
        #pragma unroll
        for (int nj = 0; nj < 4; ++nj) {
            acc[0][nj] = wmma_bf16(a0, bf[nj], acc[0][nj]);
            acc[1][nj] = wmma_bf16(a1, bf[nj], acc[1][nj]);
        }
    }

    float* Oe = Out + (size_t)e * TOK * HID;
    const int mlane = (lane >> 4) << 3;
    const int nl    = lane & 15;
    #pragma unroll
    for (int mi = 0; mi < 2; ++mi)
        #pragma unroll
        for (int nj = 0; nj < 4; ++nj)
            #pragma unroll
            for (int r = 0; r < 8; ++r) {
                int m = m0 + wm + mi * 16 + mlane + r;
                int n = n0 + wn + nj * 16 + nl;
                Oe[(size_t)m * HID + n] = acc[mi][nj][r];
            }
}

extern "C" void kernel_launch(void* const* d_in, const int* in_sizes, int n_in,
                              void* d_out, int out_size, void* d_ws, size_t ws_size,
                              hipStream_t stream) {
    (void)in_sizes; (void)n_in; (void)out_size; (void)ws_size;
    const float* X  = (const float*)d_in[0];   // (E,TOK,HID)
    const float* w1 = (const float*)d_in[1];   // (E,HID,FFN)
    const float* w2 = (const float*)d_in[2];   // (E,FFN,HID)
    const float* w3 = (const float*)d_in[3];   // (E,HID,FFN)
    float*      out = (float*)d_out;           // (E,TOK,HID)

    // ws layout (bf16 elems): Xb | W1^T | W3^T | W2^T | hidden  (~0.94 GB)
    uint16_t* ws  = (uint16_t*)d_ws;
    size_t o = 0;
    uint16_t* Xb  = ws + o; o += (size_t)NE * TOK * HID;
    uint16_t* W1t = ws + o; o += (size_t)NE * HID * FFN;   // stored as (E,FFN,HID)
    uint16_t* W3t = ws + o; o += (size_t)NE * HID * FFN;   // stored as (E,FFN,HID)
    uint16_t* W2t = ws + o; o += (size_t)NE * FFN * HID;   // stored as (E,HID,FFN)
    uint16_t* Hid = ws + o;

    dim3 blk(256);
    // one-time precision/layout pass
    int n4x = NE * TOK * HID / 4;
    convert_bf16<<<dim3(n4x / 256), blk, 0, stream>>>(X, Xb, n4x);
    convert_transpose_bf16<<<dim3(FFN / 32, HID / 32, NE), blk, 0, stream>>>(w1, W1t, HID, FFN);
    convert_transpose_bf16<<<dim3(FFN / 32, HID / 32, NE), blk, 0, stream>>>(w3, W3t, HID, FFN);
    convert_transpose_bf16<<<dim3(HID / 32, FFN / 32, NE), blk, 0, stream>>>(w2, W2t, FFN, HID);
    // WMMA GEMMs
    moe_gemm1_swiglu<<<dim3(FFN / BN, TOK / BM, NE), blk, 0, stream>>>(Xb, W1t, W3t, Hid);
    moe_gemm2      <<<dim3(HID / BN, TOK / BM, NE), blk, 0, stream>>>(Hid, W2t, out);
}
